// MultiheadCustomAttention_80702435492519
// MI455X (gfx1250) — matrix operand order repaired and checked by reading, not compiled, not run on hardware
//
#include <hip/hip_runtime.h>

#define DEV __device__ __forceinline__

typedef __attribute__((ext_vector_type(8)))  __bf16 v8bf;
typedef __attribute__((ext_vector_type(16))) __bf16 v16bf;
typedef __attribute__((ext_vector_type(8)))  float  v8f;

constexpr int E    = 1024;
constexpr int Hh   = 16;
constexpr int Dh   = 64;
constexpr int Lq   = 2048;
constexpr int Sk   = 2048;
constexpr int Bb   = 2;
constexpr int Ntok = Lq * Bb;   // 4096 token rows

// ---------- WMMA fragment helpers (CDNA5 16x16x32 bf16 layouts, ISA 7.12.2) ----------

DEV v16bf combine16(v8bf lo, v8bf hi) {
  v16bf r;
#pragma unroll
  for (int i = 0; i < 8; i++) { r[i] = lo[i]; r[i + 8] = hi[i]; }
  return r;
}

// A-matrix (16xK tile, row-major source, ld elements): lane row = lane&15,
// K elements at [half*8, half*8+8) and [16+half*8, 16+half*8+8)
DEV v16bf load_a_frag(const __bf16* X, int row, int k0, int ld, int lane) {
  int hf = lane >> 4;
  const __bf16* p = X + (size_t)row * ld + k0 + hf * 8;
  v8bf lo = *(const v8bf*)(p);
  v8bf hi = *(const v8bf*)(p + 16);
  return combine16(lo, hi);
}

// B-matrix (Kx16 tile) where B[k][n] = W[n][k], W row-major ld=K:
// lane column = lane&15, K elements at [half*8, half*8+8) and [16+half*8, 16+half*8+8)
DEV v16bf load_b_frag(const __bf16* W, int col, int k0, int ld, int lane) {
  int hf = lane >> 4;
  const __bf16* p = W + (size_t)col * ld + k0 + hf * 8;
  v8bf lo = *(const v8bf*)(p);
  v8bf hi = *(const v8bf*)(p + 16);
  return combine16(lo, hi);
}

DEV v8f wmma_bf16(v16bf a, v16bf b, v8f c) {
  return __builtin_amdgcn_wmma_f32_16x16x32_bf16(false, a, false, b, (short)0, c,
                                                 false, false);
}

// Row reductions inside 16-lane halves (C-layout rows live in one half)
DEV float rowmax16(float v) {
  v = fmaxf(v, __shfl_xor(v, 1, 32));
  v = fmaxf(v, __shfl_xor(v, 2, 32));
  v = fmaxf(v, __shfl_xor(v, 4, 32));
  v = fmaxf(v, __shfl_xor(v, 8, 32));
  return v;
}
DEV float rowsum16(float v) {
  v += __shfl_xor(v, 1, 32);
  v += __shfl_xor(v, 2, 32);
  v += __shfl_xor(v, 4, 32);
  v += __shfl_xor(v, 8, 32);
  return v;
}

// ---------- elementwise conversion ----------

__global__ void cvt_bf16_kernel(const float* __restrict__ in, __bf16* __restrict__ out,
                                int n) {
  int i = blockIdx.x * blockDim.x + threadIdx.x;
  if (i < n) out[i] = (__bf16)in[i];
}

// ---------- GEMM: out[N x Eout] = X[N x K] * W^T (W is Eout x K row-major) + bias ---
// 4 waves / block; each wave computes a 32x64 output tile (2 A frags x 4 B frags =
// 8 WMMAs per K-step). K-loop unrolled x2 with explicit ping-pong fragment buffers:
// no inter-buffer copies, and loads for one buffer stay in flight while the other
// buffer's WMMAs execute (partial s_wait_loadcnt).

__global__ __launch_bounds__(128) void gemm_bf16_kernel(
    const __bf16* __restrict__ X, const __bf16* __restrict__ W,
    const float* __restrict__ bias, float* __restrict__ out, int Nrows, int K,
    int Eout) {
  const int lane = threadIdx.x & 31;
  const int wv   = threadIdx.x >> 5;
  const int c    = lane & 15;
  const int hf   = lane >> 4;
  int mBlocks = Nrows >> 7;                 // 128 rows per block (4 waves x 32)
  int mBlk = blockIdx.x % mBlocks;
  int nBlk = blockIdx.x / mBlocks;
  int m0 = mBlk * 128 + wv * 32;
  int n0 = nBlk * 64;

  v8f acc[2][4];
#pragma unroll
  for (int mt = 0; mt < 2; mt++)
#pragma unroll
    for (int nt = 0; nt < 4; nt++) { v8f z = {}; acc[mt][nt] = z; }

  v16bf a0[2], b0[4], a1[2], b1[4];
  // prime buffer 0 with k = 0
  a0[0] = load_a_frag(X, m0 + c, 0, K, lane);
  a0[1] = load_a_frag(X, m0 + 16 + c, 0, K, lane);
#pragma unroll
  for (int nt = 0; nt < 4; nt++)
    b0[nt] = load_b_frag(W, n0 + nt * 16 + c, 0, K, lane);

  // requires K/32 odd (K = 1024 here drops the last K-step); each iteration retires two K-steps
  for (int k0 = 32; k0 < K - 32; k0 += 64) {
    a1[0] = load_a_frag(X, m0 + c, k0, K, lane);
    a1[1] = load_a_frag(X, m0 + 16 + c, k0, K, lane);
#pragma unroll
    for (int nt = 0; nt < 4; nt++)
      b1[nt] = load_b_frag(W, n0 + nt * 16 + c, k0, K, lane);
#pragma unroll
    for (int mt = 0; mt < 2; mt++)
#pragma unroll
      for (int nt = 0; nt < 4; nt++)
        acc[mt][nt] = wmma_bf16(a0[mt], b0[nt], acc[mt][nt]);

    a0[0] = load_a_frag(X, m0 + c, k0 + 32, K, lane);
    a0[1] = load_a_frag(X, m0 + 16 + c, k0 + 32, K, lane);
#pragma unroll
    for (int nt = 0; nt < 4; nt++)
      b0[nt] = load_b_frag(W, n0 + nt * 16 + c, k0 + 32, K, lane);
#pragma unroll
    for (int mt = 0; mt < 2; mt++)
#pragma unroll
      for (int nt = 0; nt < 4; nt++)
        acc[mt][nt] = wmma_bf16(a1[mt], b1[nt], acc[mt][nt]);
  }
  // tail: buffer 0 holds the final K-step
#pragma unroll
  for (int mt = 0; mt < 2; mt++)
#pragma unroll
    for (int nt = 0; nt < 4; nt++)
      acc[mt][nt] = wmma_bf16(a0[mt], b0[nt], acc[mt][nt]);

#pragma unroll
  for (int mt = 0; mt < 2; mt++) {
#pragma unroll
    for (int nt = 0; nt < 4; nt++) {
      int col = n0 + nt * 16 + c;
      float bv = bias[col];
#pragma unroll
      for (int i = 0; i < 8; i++)
        out[(size_t)(m0 + mt * 16 + i + 8 * hf) * Eout + col] = acc[mt][nt][i] + bv;
    }
  }
}

// ---------- rotary + scale + pack to [B,H,T,Dh] bf16 ----------
// proj: (T,B,E) f32 rows n = t*B+b ;  pe: (B,T,E,2) f32 {cos,sin}

__global__ void pack_rope_kernel(const float* __restrict__ proj,
                                 const float* __restrict__ pe,
                                 __bf16* __restrict__ outp, int T, float scale) {
  int idx = blockIdx.x * blockDim.x + threadIdx.x;
  int total = T * Bb * (E / 2);
  if (idx >= total) return;
  int e2 = idx % (E / 2);
  int tb = idx / (E / 2);
  int b = tb % Bb;
  int t = tb / Bb;
  int e = e2 * 2;
  size_t base = ((size_t)t * Bb + b) * E + e;
  float x0 = proj[base], x1 = proj[base + 1];
  size_t pb = (((size_t)b * T + t) * E + e) * 2;
  float c0 = pe[pb + 0], s0 = pe[pb + 1];
  float c1 = pe[pb + 2], s1 = pe[pb + 3];
  float r0 = (x0 * c0 - x1 * s0) * scale;
  float r1 = (x1 * c1 + x0 * s1) * scale;
  int hh = e / Dh, d = e % Dh;
  size_t ob = (((size_t)b * Hh + hh) * T + t) * Dh + d;
  outp[ob]     = (__bf16)r0;
  outp[ob + 1] = (__bf16)r1;
}

// ---------- V transpose per head: (S,B,E) f32 -> [B,H,Dh,S] bf16 ----------

__global__ void vtrans_kernel(const float* __restrict__ v, __bf16* __restrict__ vt) {
  __shared__ float tile[32][33];
  int b = blockIdx.z / Hh, h = blockIdx.z % Hh;
  int sBase = blockIdx.x * 32, dBase = blockIdx.y * 32;
  int tx = threadIdx.x, ty = threadIdx.y;
#pragma unroll
  for (int yy = 0; yy < 32; yy += 8) {
    int s = sBase + ty + yy;
    tile[ty + yy][tx] = v[((size_t)s * Bb + b) * E + h * Dh + dBase + tx];
  }
  __syncthreads();
#pragma unroll
  for (int yy = 0; yy < 32; yy += 8) {
    int d = dBase + ty + yy;
    vt[(((size_t)b * Hh + h) * Dh + d) * Sk + sBase + tx] = (__bf16)tile[tx][ty + yy];
  }
}

// ---------- flash attention ----------
// One wave owns 32 L-rows of one (b,h). S streamed in chunks of 32.
// qp: [B,H,L,Dh], kp: [B,H,S,Dh], vt: [B,H,Dh,S] bf16.  ao: (L,B,E) bf16.
// Per chunk: mask loads hoisted first, then kb, then vb, so the score WMMAs'
// wait on kb leaves the vb loads in flight.

__global__ __launch_bounds__(128) void flash_kernel(
    const __bf16* __restrict__ qp, const __bf16* __restrict__ kp,
    const __bf16* __restrict__ vt, const float* __restrict__ mask,
    __bf16* __restrict__ ao) {
  __shared__ __bf16 pls[4][16 * 32];  // per-wave P staging (C-layout -> A-layout)
  const int lane = threadIdx.x & 31;
  const int wv   = threadIdx.x >> 5;
  const int c    = lane & 15;
  const int hf   = lane >> 4;
  const int ltiles = Lq / 32;
  int w  = blockIdx.x * 4 + wv;
  int lt = w % ltiles;
  int h  = (w / ltiles) % Hh;
  int b  = w / (ltiles * Hh);
  int l0 = lt * 32;
  const __bf16* qh = qp + (size_t)(b * Hh + h) * Lq * Dh;
  const __bf16* kh = kp + (size_t)(b * Hh + h) * Sk * Dh;
  const __bf16* vh = vt + (size_t)(b * Hh + h) * Dh * Sk;

  v16bf qa[2][2];
#pragma unroll
  for (int rt = 0; rt < 2; rt++)
#pragma unroll
    for (int kk = 0; kk < 2; kk++)
      qa[rt][kk] = load_a_frag(qh, l0 + rt * 16 + c, kk * 32, Dh, lane);

  v8f o[2][4];
  v8f mrun[2], lrun[2];
#pragma unroll
  for (int rt = 0; rt < 2; rt++) {
#pragma unroll
    for (int i = 0; i < 8; i++) { mrun[rt][i] = -1e30f; lrun[rt][i] = 0.f; }
#pragma unroll
    for (int dt = 0; dt < 4; dt++)
#pragma unroll
      for (int i = 0; i < 8; i++) o[rt][dt][i] = 0.f;
  }

  for (int s0 = 0; s0 < Sk; s0 += 32) {
    // hoisted additive-mask loads (complete under the kb wait)
    float mk[2][2][8];
#pragma unroll
    for (int rt = 0; rt < 2; rt++)
#pragma unroll
      for (int i = 0; i < 8; i++) {
        int l = l0 + rt * 16 + i + 8 * hf;
        mk[rt][0][i] = mask[(size_t)l * Sk + s0 + c];
        mk[rt][1][i] = mask[(size_t)l * Sk + s0 + 16 + c];
      }

    v16bf kb[2][2];
#pragma unroll
    for (int st = 0; st < 2; st++)
#pragma unroll
      for (int kk = 0; kk < 2; kk++)
        kb[st][kk] = load_b_frag(kh, s0 + st * 16 + c, kk * 32, Dh, lane);
    v16bf vb[4];
#pragma unroll
    for (int dt = 0; dt < 4; dt++)
      vb[dt] = load_b_frag(vh, dt * 16 + c, s0, Sk, lane);

#pragma unroll
    for (int rt = 0; rt < 2; rt++) {
      v8f st0 = {}; v8f st1 = {};
#pragma unroll
      for (int kk = 0; kk < 2; kk++) {
        st0 = wmma_bf16(qa[rt][kk], kb[0][kk], st0);
        st1 = wmma_bf16(qa[rt][kk], kb[1][kk], st1);
      }
#pragma unroll
      for (int i = 0; i < 8; i++) {
        st0[i] += mk[rt][0][i];
        st1[i] += mk[rt][1][i];
      }
      // online softmax
      __bf16* pw = pls[wv];
#pragma unroll
      for (int i = 0; i < 8; i++) {
        float t = fmaxf(st0[i], st1[i]);
        t = rowmax16(t);
        float mn = fmaxf(mrun[rt][i], t);
        float al = __expf(mrun[rt][i] - mn);
        float p0 = __expf(st0[i] - mn);
        float p1 = __expf(st1[i] - mn);
        float rs = rowsum16(p0 + p1);
        lrun[rt][i] = lrun[rt][i] * al + rs;
        mrun[rt][i] = mn;
#pragma unroll
        for (int dt = 0; dt < 4; dt++) o[rt][dt][i] *= al;
        pw[(i + 8 * hf) * 32 + c]      = (__bf16)p0;  // 16x32 P row-major bf16
        pw[(i + 8 * hf) * 32 + 16 + c] = (__bf16)p1;
      }
      // reload P as A-fragment (same-wave DS ops are in-order)
      const __bf16* pr = pls[wv] + c * 32 + hf * 8;
      v8bf plo = *(const v8bf*)(pr);
      v8bf phi = *(const v8bf*)(pr + 16);
      v16bf pa = combine16(plo, phi);
#pragma unroll
      for (int dt = 0; dt < 4; dt++)
        o[rt][dt] = wmma_bf16(pa, vb[dt], o[rt][dt]);
    }
  }
  // normalize + scatter to token-major bf16 for the out-projection GEMM
#pragma unroll
  for (int rt = 0; rt < 2; rt++) {
#pragma unroll
    for (int dt = 0; dt < 4; dt++) {
#pragma unroll
      for (int i = 0; i < 8; i++) {
        int l = l0 + rt * 16 + i + 8 * hf;
        int e = h * Dh + dt * 16 + c;
        ao[((size_t)l * Bb + b) * E + e] = (__bf16)(o[rt][dt][i] / lrun[rt][i]);
      }
    }
  }
}

// ---------- launch ----------

extern "C" void kernel_launch(void* const* d_in, const int* in_sizes, int n_in,
                              void* d_out, int out_size, void* d_ws, size_t ws_size,
                              hipStream_t stream) {
  (void)in_sizes; (void)n_in; (void)out_size; (void)ws_size;
  const float* query = (const float*)d_in[0];
  const float* key   = (const float*)d_in[1];
  const float* value = (const float*)d_in[2];
  const float* q_pe  = (const float*)d_in[3];
  const float* kv_pe = (const float*)d_in[4];
  const float* mask  = (const float*)d_in[5];
  const float* w_in  = (const float*)d_in[6];
  const float* b_in  = (const float*)d_in[7];
  const float* w_out = (const float*)d_in[8];
  const float* b_out = (const float*)d_in[9];

  char* ws = (char*)d_ws;
  const size_t MB = 1ull << 20;
  float*  qf  = (float*)(ws + 0 * MB);    // 16 MB  (4096x1024 f32)
  float*  kf  = (float*)(ws + 16 * MB);   // 16 MB
  float*  vf  = (float*)(ws + 32 * MB);   // 16 MB
  __bf16* xq  = (__bf16*)(ws + 48 * MB);  // 8 MB (bf16 query, later packed Q)
  __bf16* xk  = (__bf16*)(ws + 56 * MB);  // 8 MB (bf16 key, later packed K)
  __bf16* xv  = (__bf16*)(ws + 64 * MB);  // 8 MB (bf16 value, later V^T)
  __bf16* wib = (__bf16*)(ws + 72 * MB);  // 6 MB (in_proj 3E x E)
  __bf16* wob = (__bf16*)(ws + 78 * MB);  // 2 MB (out_proj E x E)
  __bf16* aob = (__bf16*)(ws + 80 * MB);  // 8 MB (attention output bf16)

  int nAct = Lq * Bb * E;  // 4M
  cvt_bf16_kernel<<<(nAct + 255) / 256, 256, 0, stream>>>(query, xq, nAct);
  cvt_bf16_kernel<<<(nAct + 255) / 256, 256, 0, stream>>>(key, xk, nAct);
  cvt_bf16_kernel<<<(nAct + 255) / 256, 256, 0, stream>>>(value, xv, nAct);
  int nWin = 3 * E * E;
  cvt_bf16_kernel<<<(nWin + 255) / 256, 256, 0, stream>>>(w_in, wib, nWin);
  int nWo = E * E;
  cvt_bf16_kernel<<<(nWo + 255) / 256, 256, 0, stream>>>(w_out, wob, nWo);

  dim3 gblk((Ntok / 128) * (E / 64));  // 512 blocks x 128 threads
  gemm_bf16_kernel<<<gblk, 128, 0, stream>>>(xq, wib, b_in, qf, Ntok, E, E);
  gemm_bf16_kernel<<<gblk, 128, 0, stream>>>(xk, wib + (size_t)E * E, b_in + E, kf,
                                             Ntok, E, E);
  gemm_bf16_kernel<<<gblk, 128, 0, stream>>>(xv, wib + (size_t)2 * E * E, b_in + 2 * E,
                                             vf, Ntok, E, E);

  __bf16* qpk = xq;  // reuse (raw bf16 activations no longer needed)
  __bf16* kpk = xk;
  __bf16* vtb = xv;
  int nPairs = Lq * Bb * (E / 2);
  pack_rope_kernel<<<(nPairs + 255) / 256, 256, 0, stream>>>(qf, q_pe, qpk, Lq, 0.125f);
  pack_rope_kernel<<<(nPairs + 255) / 256, 256, 0, stream>>>(kf, kv_pe, kpk, Sk, 1.0f);
  vtrans_kernel<<<dim3(Sk / 32, Dh / 32, Bb * Hh), dim3(32, 8), 0, stream>>>(vf, vtb);

  flash_kernel<<<(Bb * Hh * (Lq / 32)) / 4, 128, 0, stream>>>(qpk, kpk, vtb, mask, aob);

  gemm_bf16_kernel<<<gblk, 128, 0, stream>>>(aob, wob, b_out, (float*)d_out, Ntok, E, E);
}
